// SNNModel_63745904608019
// MI455X (gfx1250) — matrix operand order repaired and checked by reading, not compiled, not run on hardware
//
#include <hip/hip_runtime.h>
#include <hip/hip_bf16.h>
#include <stdint.h>

// ---------------------------------------------------------------------------
// MI455X (gfx1250) SNN forward pass.
//
// Strategy (see analysis): weights converted to bf16 (288MB -> 144MB, fits
// 192MB L2), pre-swizzled into a contiguous-k (k = 16h + s) v_wmma_f32_16x16x32_bf16 A-matrix
// lane layout, and every per-step mat-vec runs through the WMMA pipe with the
// input vector broadcast across all 16 B columns. 5 kernel launches per time
// step, 999 steps, all on `stream` (graph-capture safe).
// ---------------------------------------------------------------------------

typedef __bf16 bf16_t;
typedef __attribute__((ext_vector_type(16))) __bf16 v16bf;
typedef __attribute__((ext_vector_type(8)))  float  v8f;

#define KDIM   4096   // reduction length of every mat-vec in this model
#define NSTEPS 1000   // T
#define TCOLS  999    // T-1 output columns

struct LayerArgs {
  const bf16_t* Wt;           // tiled bf16 weights (WMMA A layout)
  const bf16_t* vecDirect;    // if non-null: direct bf16 input vector [KDIM]
  const bf16_t* buf1;         // circular spike buffer, slot-major [16][splitN]
  const bf16_t* buf2;         // second buffer for concatenated (CN) input
  const int*    delays;       // per-presynaptic delay [KDIM]
  const float*  inhVec;       // per-row inhibition vector (or null)
  const float*  inhScalarPtr; // scalar inhibition (or null)
  float*        Vst;          // membrane potential state [nRows]
  float*        spikeVec;     // optional: write current spikes [nRows]
  float*        sumAccum;     // optional: atomicAdd spike count
  bf16_t*       pushBuf;      // optional: push spikes into circular buffer
  float         scale;        // current scale (1.0 or 2.0)
  float         inhCoef;      // coefficient for inhVec
  int           splitN;       // neurons in buf1 (KDIM if single buffer)
  int           nRows;        // output neurons of this layer
  int           outBase;      // row offset into d_out
};

// ---------------------------------------------------------------------------
// Hot kernel: LIF layer update.  256 threads = 8 waves; each wave produces 16
// output neurons via a K-loop of 128 WMMA 16x16x32 bf16 tiles.
// Grid is split: blocks [0, blocksA0) use A0, the rest use A1.
// ---------------------------------------------------------------------------
__global__ __launch_bounds__(256) void lif_layer_kernel(
    LayerArgs A0, LayerArgs A1, int blocksA0,
    int pos, int tcol, float* __restrict__ out)
{
  __shared__ __align__(32) bf16_t ldsV[KDIM];

  const bool first = ((int)blockIdx.x < blocksA0);
  const LayerArgs P = first ? A0 : A1;
  const int blk = first ? (int)blockIdx.x : ((int)blockIdx.x - blocksA0);
  const int tid = threadIdx.x;

  // Stage the B vector (bf16) into LDS: either the direct input for this
  // time step, or the per-neuron delayed spikes gathered from the circular
  // buffer(s): delayed[j] = buf[(pos - d[j]) & 15][j].
  if (P.vecDirect) {
    for (int j = tid; j < KDIM; j += 256) ldsV[j] = P.vecDirect[j];
  } else {
    for (int j = tid; j < KDIM; j += 256) {
      const int slot = (pos - P.delays[j]) & 15;
      if (j < P.splitN)
        ldsV[j] = P.buf1[slot * P.splitN + j];
      else
        ldsV[j] = P.buf2[slot * (KDIM - P.splitN) + (j - P.splitN)];
    }
  }
  __syncthreads();

  const int wave    = tid >> 5;
  const int lane    = tid & 31;
  const int rowTile = blk * 8 + wave;     // 16 rows per wave, 128 per block

  // Tiled weights: tile (rowTile, kt) is 1KB = 32 lanes x 16 bf16, already in
  // a contiguous-k A layout (lane half h holds k 16h..16h+15), so each lane loads 32 contiguous bytes.
  const v16bf* wptr = reinterpret_cast<const v16bf*>(P.Wt)
                      + (size_t)rowTile * (128 * 32) + lane;
  // B operand broadcast: lanes 0-15 need K[kt*32 .. +16), lanes 16-31 the
  // next 16 (all 16 columns identical = the input vector chunk).
  const v16bf* bptr = reinterpret_cast<const v16bf*>(ldsV) + (lane >> 4);

  v8f acc = {};
  #pragma unroll 4
  for (int kt = 0; kt < 128; ++kt) {
    v16bf a = wptr[kt * 32];
    v16bf b = bptr[kt * 2];
    acc = __builtin_amdgcn_wmma_f32_16x16x32_bf16(
        /*neg_a=*/false, a, /*neg_b=*/false, b,
        /*c_mod=*/(short)0, acc, /*reuse_a=*/false, /*reuse_b=*/false);
  }

  // All D columns are identical; lane 0 holds rows M=0..7 in acc[0..7],
  // lane 16 holds rows M=8..15.  LIF update + all side effects.
  if (lane == 0 || lane == 16) {
    const int mOff = (lane == 16) ? 8 : 0;
    #pragma unroll
    for (int r = 0; r < 8; ++r) {
      const int row = rowTile * 16 + mOff + r;
      float I = acc[r] * P.scale;
      if (P.inhVec)       I -= P.inhVec[row] * P.inhCoef;
      if (P.inhScalarPtr) I -= *P.inhScalarPtr;
      float v = P.Vst[row] * 0.9f + I;
      const float s = (v >= 1.0f) ? 1.0f : 0.0f;
      P.Vst[row] = v * (1.0f - s);
      out[(size_t)(P.outBase + row) * TCOLS + tcol] = s;
      if (P.pushBuf)  P.pushBuf[pos * P.nRows + row] = (bf16_t)s;
      if (P.spikeVec) P.spikeVec[row] = s;
      if (P.sumAccum && s > 0.0f) atomicAdd(P.sumAccum, s);  // exact: sums of 1.0
    }
  }
}

// ---------------------------------------------------------------------------
// One-time per call: convert f32 weights -> bf16 in the WMMA A-matrix tile
// layout.  Tiled element e: tile = e/512, lane = (e/16)%32, j = e%16.
//   lane<16 : row = rt*16 + lane,      K = kt*32 + j
//   lane>=16: row = rt*16 + lane - 16, K = kt*32 + 16 + j
// ---------------------------------------------------------------------------
__global__ __launch_bounds__(256) void tile_weights_kernel(
    const float* __restrict__ W, bf16_t* __restrict__ Wt, int nRows)
{
  const size_t idx   = (size_t)blockIdx.x * 256 + threadIdx.x;
  const size_t total = (size_t)nRows * KDIM / 2;
  if (idx >= total) return;
  const size_t e    = idx * 2;              // j is even -> pair stays in region
  const int    j    = (int)(e & 15);
  const int    lane = (int)((e >> 4) & 31);
  const size_t tile = e >> 9;
  const int    kt   = (int)(tile & 127);
  const int    rt   = (int)(tile >> 7);
  const int    row  = rt * 16 + (lane & 15);
  const int    hi   = lane >> 4;
  const int    kk   = hi * 16 + j;
  const size_t src  = (size_t)row * KDIM + (size_t)kt * 32 + kk;
  Wt[e]     = (bf16_t)W[src];
  Wt[e + 1] = (bf16_t)W[src + 1];
}

// ---------------------------------------------------------------------------
// One-time per call: transpose stim -> time-major bf16 SA/RA input vectors.
// saT[t][p] = stim[p][t];  raT[t][p] = |stim[p][t]-stim[p][t-1]| * 30.
// ---------------------------------------------------------------------------
__global__ __launch_bounds__(256) void prep_inputs_kernel(
    const float* __restrict__ stim,
    bf16_t* __restrict__ saT, bf16_t* __restrict__ raT)
{
  const size_t idx = (size_t)blockIdx.x * 256 + threadIdx.x;
  if (idx >= (size_t)KDIM * NSTEPS) return;
  const int t = (int)(idx % NSTEPS);
  const int p = (int)(idx / NSTEPS);
  const float cur = stim[(size_t)p * NSTEPS + t];
  saT[(size_t)t * KDIM + p] = (bf16_t)cur;
  float ra = 0.0f;
  if (t > 0) ra = fabsf(cur - stim[(size_t)p * NSTEPS + t - 1]) * 30.0f;
  raT[(size_t)t * KDIM + p] = (bf16_t)ra;
}

__global__ __launch_bounds__(256) void zero_kernel(uint32_t* __restrict__ p, size_t nWords)
{
  const size_t i = (size_t)blockIdx.x * 256 + threadIdx.x;
  if (i < nWords) p[i] = 0u;
}

// ---------------------------------------------------------------------------
extern "C" void kernel_launch(void* const* d_in, const int* in_sizes, int n_in,
                              void* d_out, int out_size, void* d_ws, size_t ws_size,
                              hipStream_t stream)
{
  (void)in_sizes; (void)n_in; (void)out_size;

  const float* stim = (const float*)d_in[0];
  const float* Wsa0 = (const float*)d_in[1];
  const float* Wsa1 = (const float*)d_in[2];
  const float* Wsa2 = (const float*)d_in[3];
  const float* Wra0 = (const float*)d_in[4];
  const float* Wra1 = (const float*)d_in[5];
  const float* Wra2 = (const float*)d_in[6];
  const float* Wcn0 = (const float*)d_in[7];
  const float* Wcn1 = (const float*)d_in[8];
  const int* d_sa1 = (const int*)d_in[9];
  const int* d_sa2 = (const int*)d_in[10];
  const int* d_ra1 = (const int*)d_in[11];
  const int* d_ra2 = (const int*)d_in[12];
  const int* d_cn0 = (const int*)d_in[13];
  const int* d_cn1 = (const int*)d_in[14];
  float* out = (float*)d_out;

  const int N0 = 4096, N1 = 2048, N2 = 2048, NCN = 1024;

  // ---- workspace layout -------------------------------------------------
  char* base = (char*)d_ws;
  size_t off = 0;
  auto take = [&](size_t bytes) -> char* {
    char* p = base + off;
    off = (off + bytes + 255) & ~(size_t)255;
    return p;
  };

  bf16_t* Wt_sa0 = (bf16_t*)take((size_t)N0  * KDIM * 2);
  bf16_t* Wt_sa1 = (bf16_t*)take((size_t)N1  * KDIM * 2);
  bf16_t* Wt_sa2 = (bf16_t*)take((size_t)N2  * KDIM * 2);
  bf16_t* Wt_ra0 = (bf16_t*)take((size_t)N0  * KDIM * 2);
  bf16_t* Wt_ra1 = (bf16_t*)take((size_t)N1  * KDIM * 2);
  bf16_t* Wt_ra2 = (bf16_t*)take((size_t)N2  * KDIM * 2);
  bf16_t* Wt_cn0 = (bf16_t*)take((size_t)NCN * KDIM * 2);
  bf16_t* Wt_cn1 = (bf16_t*)take((size_t)NCN * KDIM * 2);
  bf16_t* saT    = (bf16_t*)take((size_t)NSTEPS * KDIM * 2);
  bf16_t* raT    = (bf16_t*)take((size_t)NSTEPS * KDIM * 2);

  const size_t stateBeg = off;          // everything below is zero-initialized
  bf16_t* b_s0 = (bf16_t*)take((size_t)16 * N0 * 2);
  bf16_t* b_r0 = (bf16_t*)take((size_t)16 * N0 * 2);
  bf16_t* b_s1 = (bf16_t*)take((size_t)16 * N1 * 2);
  bf16_t* b_r1 = (bf16_t*)take((size_t)16 * N1 * 2);
  float* v_s0 = (float*)take((size_t)N0  * 4);
  float* v_r0 = (float*)take((size_t)N0  * 4);
  float* v_s1 = (float*)take((size_t)N1  * 4);
  float* v_s2 = (float*)take((size_t)N2  * 4);
  float* v_r1 = (float*)take((size_t)N1  * 4);
  float* v_r2 = (float*)take((size_t)N2  * 4);
  float* v_c0 = (float*)take((size_t)NCN * 4);
  float* v_c1 = (float*)take((size_t)NCN * 4);
  float* sS2   = (float*)take((size_t)N2  * 4);
  float* sC1   = (float*)take((size_t)NCN * 4);
  float* sumR2 = (float*)take((size_t)NSTEPS * 4);
  const size_t stateEnd = off;

  if (ws_size < off) return;  // workspace too small: cannot run safely

  // ---- one-time per call: init state, tile weights, prep inputs ---------
  {
    const size_t nWords = (stateEnd - stateBeg) / 4;
    zero_kernel<<<(unsigned)((nWords + 255) / 256), 256, 0, stream>>>(
        (uint32_t*)(base + stateBeg), nWords);
  }
  auto tileW = [&](const float* W, bf16_t* Wt, int n) {
    const size_t total = (size_t)n * KDIM / 2;
    tile_weights_kernel<<<(unsigned)((total + 255) / 256), 256, 0, stream>>>(W, Wt, n);
  };
  tileW(Wsa0, Wt_sa0, N0);  tileW(Wsa1, Wt_sa1, N1);  tileW(Wsa2, Wt_sa2, N2);
  tileW(Wra0, Wt_ra0, N0);  tileW(Wra1, Wt_ra1, N1);  tileW(Wra2, Wt_ra2, N2);
  tileW(Wcn0, Wt_cn0, NCN); tileW(Wcn1, Wt_cn1, NCN);
  {
    const size_t total = (size_t)KDIM * NSTEPS;
    prep_inputs_kernel<<<(unsigned)((total + 255) / 256), 256, 0, stream>>>(stim, saT, raT);
  }

  // ---- time loop: 5 stream-ordered launches per step ---------------------
  for (int t = 1; t < NSTEPS; ++t) {
    const int pos = t & 15, tcol = t - 1;

    // L0: SA0 + RA0 (direct stimulus, push level-0 buffers)
    LayerArgs a0{}, a1{};
    a0.Wt = Wt_sa0; a0.vecDirect = saT + (size_t)t * KDIM; a0.Vst = v_s0;
    a0.pushBuf = b_s0; a0.scale = 1.0f; a0.splitN = KDIM; a0.nRows = N0; a0.outBase = 0;
    a1 = a0;
    a1.Wt = Wt_ra0; a1.vecDirect = raT + (size_t)t * KDIM; a1.Vst = v_r0;
    a1.pushBuf = b_r0; a1.outBase = 8192;
    lif_layer_kernel<<<64, 256, 0, stream>>>(a0, a1, 32, pos, tcol, out);

    // L2: SA2 (records sS2) + RA2 (records sum of spikes)
    LayerArgs s2{}, r2{};
    s2.Wt = Wt_sa2; s2.buf1 = b_s0; s2.splitN = KDIM; s2.delays = d_sa2;
    s2.scale = 2.0f; s2.Vst = v_s2; s2.spikeVec = sS2; s2.nRows = N2; s2.outBase = 6144;
    r2.Wt = Wt_ra2; r2.buf1 = b_r0; r2.splitN = KDIM; r2.delays = d_ra2;
    r2.scale = 2.0f; r2.Vst = v_r2; r2.sumAccum = sumR2 + t; r2.nRows = N2; r2.outBase = 14336;
    lif_layer_kernel<<<32, 256, 0, stream>>>(s2, r2, 16, pos, tcol, out);

    // L1: SA1 (per-row inhibition by sS2*20) + RA1 (scalar inhibition), push PN buffers
    LayerArgs s1{}, r1{};
    s1.Wt = Wt_sa1; s1.buf1 = b_s0; s1.splitN = KDIM; s1.delays = d_sa1;
    s1.scale = 2.0f; s1.inhVec = sS2; s1.inhCoef = 20.0f; s1.Vst = v_s1;
    s1.pushBuf = b_s1; s1.nRows = N1; s1.outBase = 4096;
    r1.Wt = Wt_ra1; r1.buf1 = b_r0; r1.splitN = KDIM; r1.delays = d_ra1;
    r1.scale = 2.0f; r1.inhScalarPtr = sumR2 + t; r1.Vst = v_r1;
    r1.pushBuf = b_r1; r1.nRows = N1; r1.outBase = 12288;
    lif_layer_kernel<<<32, 256, 0, stream>>>(s1, r1, 16, pos, tcol, out);

    // CN1: reads concatenated [b_s1 ; b_r1]
    LayerArgs c1{};
    c1.Wt = Wt_cn1; c1.buf1 = b_s1; c1.buf2 = b_r1; c1.splitN = N1; c1.delays = d_cn1;
    c1.scale = 1.0f; c1.Vst = v_c1; c1.spikeVec = sC1; c1.nRows = NCN; c1.outBase = 17408;
    lif_layer_kernel<<<8, 256, 0, stream>>>(c1, c1, 8, pos, tcol, out);

    // CN0: same PN input, inhibited by sC1*10
    LayerArgs c0{};
    c0.Wt = Wt_cn0; c0.buf1 = b_s1; c0.buf2 = b_r1; c0.splitN = N1; c0.delays = d_cn0;
    c0.scale = 1.0f; c0.inhVec = sC1; c0.inhCoef = 10.0f; c0.Vst = v_c0;
    c0.nRows = NCN; c0.outBase = 16384;
    lif_layer_kernel<<<8, 256, 0, stream>>>(c0, c0, 8, pos, tcol, out);
  }
}
